// SwitchMoE_8881992368572
// MI455X (gfx1250) — matrix operand order repaired and checked by reading, not compiled, not run on hardware
//
#include <hip/hip_runtime.h>

#define DEV __device__ __forceinline__

typedef __attribute__((ext_vector_type(16))) __bf16        v16bf;
typedef __attribute__((ext_vector_type(8)))  float         v8f;
typedef __attribute__((ext_vector_type(4)))  unsigned int  v4u;
typedef __attribute__((ext_vector_type(2)))  unsigned int  v2u;
typedef unsigned short us;

constexpr int D  = 1024;     // hidden dim
constexpr int FF = 4096;     // MLP hidden
constexpr int E  = 8;        // experts
constexpr int T  = 16384;    // tokens (B*S)
constexpr int MT = 64;       // tokens per MLP block (4 row-tiles of 16)
constexpr int RT = MT / 16;  // row tiles
constexpr int FCHUNK = 256;  // FF columns per fused chunk (8 waves x 32)
constexpr int KS = 64;       // K rows staged per phase-A step

// ---- f32 -> bf16 via native __bf16 (hardware cvt; pairs fuse to cvt_pk) ------
DEV us bf_bits(float f) {
  __bf16 h = (__bf16)f;
  return __builtin_bit_cast(us, h);
}
DEV unsigned pk_bf16(float a, float b) {   // low = a, high = b
  return ((unsigned)bf_bits(b) << 16) | bf_bits(a);
}

DEV v8f wmma_bf16(v16bf a, v16bf b, v8f c) {
  return __builtin_amdgcn_wmma_f32_16x16x32_bf16(
      false, a, false, b, (short)0, c, false, false);
}

// A-fragment (16x32 bf16) from row-major bf16(bits) LDS tile: two b128 loads.
DEV v16bf a_frag(const us* p, int ld, int k0, int lane) {
  int M = lane & 15, half = lane >> 4;
  int base = M * ld + k0 + half * 16;
  union { v4u q[2]; v16bf v; } u;
  u.q[0] = *reinterpret_cast<const v4u*>(p + base);
  u.q[1] = *reinterpret_cast<const v4u*>(p + base + 8);
  return u.v;
}

// B-fragment (32x16 bf16) from fragment-order LDS staging: two b128 loads.
// Staging layout: tile*512 + lane*16 + i, where lane = (N | (K>=16)<<4), i = K&15.
DEV v16bf b_frag(const us* wst, int tile, int lane) {
  const us* p = wst + tile * 512 + lane * 16;
  union { v4u q[2]; v16bf v; } u;
  u.q[0] = *reinterpret_cast<const v4u*>(p);
  u.q[1] = *reinterpret_cast<const v4u*>(p + 8);
  return u.v;
}

// ============================ kernel 1: init ==================================
__global__ __launch_bounds__(256) void moe_init(float* __restrict__ out,
                                                int* __restrict__ cnt) {
  size_t i = (size_t)blockIdx.x * 256 + threadIdx.x;
  v4u z = {0u, 0u, 0u, 0u};
  reinterpret_cast<v4u*>(out)[i] = z;
  if (blockIdx.x == 0 && threadIdx.x < E) cnt[threadIdx.x] = 0;
}

// ============================ kernel 2: routing ===============================
// softmax is monotonic and only top-2 membership matters -> top-2 of logits.
__global__ __launch_bounds__(256) void moe_route(const float* __restrict__ x,
                                                 const float* __restrict__ Wg,
                                                 const float* __restrict__ bg,
                                                 int* __restrict__ cnt,
                                                 int* __restrict__ ids) {
  __shared__ float wg_s[D * E];  // 32 KB
  for (int i = threadIdx.x; i < D * E; i += 256) wg_s[i] = Wg[i];
  __syncthreads();

  int t = blockIdx.x * 256 + threadIdx.x;
  const float4* xr = reinterpret_cast<const float4*>(x + (size_t)t * D);
  float l[E];
#pragma unroll
  for (int e = 0; e < E; ++e) l[e] = bg[e];
#pragma unroll 1
  for (int i = 0; i < D / 4; ++i) {
    float4 v = xr[i];
    const float* w = &wg_s[i * 4 * E];
#pragma unroll
    for (int e = 0; e < E; ++e)
      l[e] += v.x * w[e] + v.y * w[E + e] + v.z * w[2 * E + e] + v.w * w[3 * E + e];
  }
  int e1 = 0; float m1 = l[0];
#pragma unroll
  for (int e = 1; e < E; ++e) if (l[e] > m1) { m1 = l[e]; e1 = e; }
  int e2 = (e1 == 0) ? 1 : 0; float m2 = -3.4e38f;
#pragma unroll
  for (int e = 0; e < E; ++e) if (e != e1 && l[e] > m2) { m2 = l[e]; e2 = e; }

  int s1 = atomicAdd(&cnt[e1], 1); ids[e1 * T + s1] = t;
  int s2 = atomicAdd(&cnt[e2], 1); ids[e2 * T + s2] = t;
}

// ============================ kernel 3: fused expert MLP ======================
// Block = 64 gathered tokens x 1 expert; 8 waves; dynamic LDS ~224 KB:
//   xs  [64 x 1024] bf16 (128 KB)  activations
//   hs  [64 x  256] bf16 ( 32 KB)  relu(h) chunk
//   wst [32 x 1024] bf16 ( 64 KB)  fragment-order staged weights
__global__ __launch_bounds__(256) void moe_mlp(const float* __restrict__ x,
                                               const float* __restrict__ W1,
                                               const float* __restrict__ b1,
                                               const float* __restrict__ W2,
                                               const float* __restrict__ b2,
                                               float* __restrict__ out,
                                               const int* __restrict__ cnt,
                                               const int* __restrict__ ids) {
  extern __shared__ char smem[];
  us*  xs   = reinterpret_cast<us*>(smem);          // MT*D
  us*  hs   = xs + MT * D;                          // MT*FCHUNK
  us*  wst  = hs + MT * FCHUNK;                     // 32*1024
  int* toks = reinterpret_cast<int*>(wst + 32 * 1024);

  const int e = blockIdx.y;
  const int count = cnt[e];
  const int m0 = blockIdx.x * MT;
  if (m0 >= count) return;

  const int tid = threadIdx.x;
  const int lane = tid & 31, wave = tid >> 5;

  if (tid < MT) {
    int row = m0 + tid;
    toks[tid] = (row < count) ? ids[e * T + row] : -1;
  }
  __syncthreads();

  // ---- gather token rows -> bf16 LDS (coalesced float4, packed b64 stores)
#pragma unroll 1
  for (int it = 0; it < MT * (D / 4) / 256; ++it) {
    int idx = it * 256 + tid;
    int row = idx >> 8;                       // D/4 = 256 float4 per row
    int c4  = idx & 255;
    int tok = toks[row]; if (tok < 0) tok = 0;
    float4 v = reinterpret_cast<const float4*>(x + (size_t)tok * D)[c4];
    v2u w = {pk_bf16(v.x, v.y), pk_bf16(v.z, v.w)};
    *reinterpret_cast<v2u*>(&xs[row * D + c4 * 4]) = w;
  }
  __syncthreads();

  const float* W1e = W1 + (size_t)e * D * FF;
  const float* b1e = b1 + (size_t)e * FF;
  const float* W2e = W2 + (size_t)e * FF * D;
  const float* b2e = b2 + (size_t)e * D;

  const v8f vz = {0.f, 0.f, 0.f, 0.f, 0.f, 0.f, 0.f, 0.f};
  v8f acc[RT][8];                             // 64 rows x 128 cols per wave
#pragma unroll
  for (int rt = 0; rt < RT; ++rt)
#pragma unroll
    for (int j = 0; j < 8; ++j) acc[rt][j] = vz;

  const int N = lane & 15, half = lane >> 4;

#pragma unroll 1
  for (int ff0 = 0; ff0 < FF; ff0 += FCHUNK) {
    // ======== phase A: h[:, ff0 + wave*32 .. +31] = relu(x @ W1 + b1) ========
    v8f ha[RT][2];
#pragma unroll
    for (int rt = 0; rt < RT; ++rt) { ha[rt][0] = vz; ha[rt][1] = vz; }

#pragma unroll 1
    for (int ks = 0; ks < D / KS; ++ks) {
      // stage W1[ks*64 .. +64][ff0 .. +256] into fragment order (bf16)
      const float* Wb = W1e + (size_t)(ks * KS) * FF + ff0;
#pragma unroll
      for (int un = 0; un < 4; ++un) {                 // 1024 units / 256 thr
        int u  = un * 256 + tid;
        int kl = (u >> 6) * 4;                         // k_local 0..60
        int c  = (u & 63) * 4;                         // col 0..252
        float4 r0 = *reinterpret_cast<const float4*>(Wb + (size_t)(kl + 0) * FF + c);
        float4 r1 = *reinterpret_cast<const float4*>(Wb + (size_t)(kl + 1) * FF + c);
        float4 r2 = *reinterpret_cast<const float4*>(Wb + (size_t)(kl + 2) * FF + c);
        float4 r3 = *reinterpret_cast<const float4*>(Wb + (size_t)(kl + 3) * FF + c);
        const float* p0 = &r0.x; const float* p1 = &r1.x;
        const float* p2 = &r2.x; const float* p3 = &r3.x;
        int ksub = kl >> 5, krem = kl & 31;
#pragma unroll
        for (int cc = 0; cc < 4; ++cc) {
          int col  = c + cc;
          int ln   = (col & 15) + ((krem >= 16) ? 16 : 0);
          int tile = ksub * 16 + (col >> 4);
          v2u w = {pk_bf16(p0[cc], p1[cc]), pk_bf16(p2[cc], p3[cc])};
          *reinterpret_cast<v2u*>(&wst[tile * 512 + ln * 16 + (kl & 15)]) = w;
        }
      }
      __syncthreads();

#pragma unroll
      for (int ksub = 0; ksub < 2; ++ksub) {
        int k0 = ks * KS + ksub * 32;
        v16bf a[RT];
#pragma unroll
        for (int rt = 0; rt < RT; ++rt) a[rt] = a_frag(xs + rt * 16 * D, D, k0, lane);
#pragma unroll
        for (int t = 0; t < 2; ++t) {
          v16bf b = b_frag(wst, ksub * 16 + wave * 2 + t, lane);
#pragma unroll
          for (int rt = 0; rt < RT; ++rt) ha[rt][t] = wmma_bf16(a[rt], b, ha[rt][t]);
        }
      }
      __syncthreads();                                 // wst reused next step
    }

    // bias + relu -> bf16 h tile in LDS
#pragma unroll
    for (int t = 0; t < 2; ++t) {
      int col = wave * 32 + t * 16 + N;
      float bias = b1e[ff0 + col];
#pragma unroll
      for (int rt = 0; rt < RT; ++rt)
#pragma unroll
        for (int r = 0; r < 8; ++r) {
          int row = rt * 16 + r + half * 8;            // C layout: M = r + 8*half
          hs[row * FCHUNK + col] = bf_bits(fmaxf(ha[rt][t][r] + bias, 0.0f));
        }
    }

    // ======== phase B: y(64x1024) += relu_h(64x256) @ W2[chunk, :] ==========
#pragma unroll 1
    for (int kk = 0; kk < FCHUNK / 32; ++kk) {
      // stage W2[ff0+kk*32 .. +32][0..1024] into fragment order (bf16)
      const float* Wb = W2e + (size_t)(ff0 + kk * 32) * D;
      if (kk + 1 < FCHUNK / 32)
        __builtin_prefetch(Wb + (size_t)32 * D + tid * 16, 0, 1);
#pragma unroll
      for (int un = 0; un < 8; ++un) {                 // 2048 units / 256 thr
        int u  = un * 256 + tid;
        int kl = (u >> 8) * 4;                         // k_local 0..28
        int c  = (u & 255) * 4;                        // col 0..1020
        float4 r0 = *reinterpret_cast<const float4*>(Wb + (size_t)(kl + 0) * D + c);
        float4 r1 = *reinterpret_cast<const float4*>(Wb + (size_t)(kl + 1) * D + c);
        float4 r2 = *reinterpret_cast<const float4*>(Wb + (size_t)(kl + 2) * D + c);
        float4 r3 = *reinterpret_cast<const float4*>(Wb + (size_t)(kl + 3) * D + c);
        const float* p0 = &r0.x; const float* p1 = &r1.x;
        const float* p2 = &r2.x; const float* p3 = &r3.x;
#pragma unroll
        for (int cc = 0; cc < 4; ++cc) {
          int col  = c + cc;
          int ln   = (col & 15) + ((kl >= 16) ? 16 : 0);
          int tile = col >> 4;
          v2u w = {pk_bf16(p0[cc], p1[cc]), pk_bf16(p2[cc], p3[cc])};
          *reinterpret_cast<v2u*>(&wst[tile * 512 + ln * 16 + (kl & 15)]) = w;
        }
      }
      __syncthreads();     // also covers hs writes on first kk

      v16bf a[RT];
#pragma unroll
      for (int rt = 0; rt < RT; ++rt)
        a[rt] = a_frag(hs + rt * 16 * FCHUNK, FCHUNK, kk * 32, lane);
#pragma unroll
      for (int j = 0; j < 8; ++j) {
        v16bf b = b_frag(wst, wave * 8 + j, lane);
#pragma unroll
        for (int rt = 0; rt < RT; ++rt) acc[rt][j] = wmma_bf16(a[rt], b, acc[rt][j]);
      }
      __syncthreads();                                 // wst/hs reuse
    }
  }

  // ---- scatter: out[token] += y + b2 (each token hit by exactly 2 experts)
  int trow[RT][8];
#pragma unroll
  for (int rt = 0; rt < RT; ++rt)
#pragma unroll
    for (int r = 0; r < 8; ++r) trow[rt][r] = toks[rt * 16 + r + half * 8];

#pragma unroll
  for (int j = 0; j < 8; ++j) {
    int ncol = wave * 128 + j * 16 + N;
    float bias = b2e[ncol];
#pragma unroll
    for (int rt = 0; rt < RT; ++rt)
#pragma unroll
      for (int r = 0; r < 8; ++r)
        if (trow[rt][r] >= 0)
          atomicAdd(out + (size_t)trow[rt][r] * D + ncol, acc[rt][j][r] + bias);
  }
}

// ============================ launcher =======================================
extern "C" void kernel_launch(void* const* d_in, const int* in_sizes, int n_in,
                              void* d_out, int out_size, void* d_ws, size_t ws_size,
                              hipStream_t stream) {
  (void)in_sizes; (void)n_in; (void)out_size; (void)ws_size;
  const float* x  = (const float*)d_in[0];
  const float* Wg = (const float*)d_in[1];
  const float* bg = (const float*)d_in[2];
  const float* W1 = (const float*)d_in[3];
  const float* b1 = (const float*)d_in[4];
  const float* W2 = (const float*)d_in[5];
  const float* b2 = (const float*)d_in[6];
  float* out = (float*)d_out;

  int* cnt = (int*)d_ws;                      // E counters
  int* ids = (int*)((char*)d_ws + 64);        // E*T compacted token ids

  constexpr size_t LDS_BYTES =
      (size_t)(MT * D + MT * FCHUNK + 32 * 1024) * sizeof(us) + MT * sizeof(int);

  moe_init<<<(T * D / 4) / 256, 256, 0, stream>>>(out, cnt);
  moe_route<<<T / 256, 256, 0, stream>>>(x, Wg, bg, cnt, ids);
  moe_mlp<<<dim3(T / MT, E), 256, LDS_BYTES, stream>>>(x, W1, b1, W2, b2, out,
                                                       cnt, ids);
}
